// OriTripletLoss_77309412139
// MI455X (gfx1250) — matrix operand order repaired and checked by reading, not compiled, not run on hardware
//
#include <hip/hip_runtime.h>
#include <hip/hip_bf16.h>
#include <stdint.h>

#define N_ROWS 4096
#define K_DIM  2048
#define TM 128
#define TN 128
#define TK 32
#define AS_STRIDE 40    // u16 per A-tile row in LDS (32 + 8 pad)
#define BS_STRIDE 136   // u16 per B-tile row in LDS (128 + 8 pad)
#define SPLIT_T 64      // split/transpose tile edge
#define SPLIT_PAD 72    // LDS stride for transpose tile

typedef __attribute__((ext_vector_type(16))) __bf16 v16bf;
typedef __attribute__((ext_vector_type(8)))  float  v8f;

union BF16Frag { v16bf v; uint4 u[2]; };

// Truncating fp32 -> bf16 split: x = hi + lo, |x - hi - lo| ~ 2^-16 * |x|.
__device__ inline unsigned short bf16_hi_trunc(float f) {
    return (unsigned short)(__float_as_uint(f) >> 16);
}
__device__ inline float bf16_as_f32(unsigned short h) {
    return __uint_as_float(((unsigned int)h) << 16);
}
__device__ inline void split_bf16(float f, unsigned short& hi, unsigned short& lo) {
    hi = bf16_hi_trunc(f);
    float r = f - bf16_as_f32(hi);
    lo = bf16_hi_trunc(r);
}

// ---------------------------------------------------------------------------
// Kernel 1a: one-time fp32 -> bf16 hi/lo split.
//  Ahi/Alo: N x K row-major; BhiT/BloT: K x N transposed (LDS-tiled transpose).
// ---------------------------------------------------------------------------
__global__ __launch_bounds__(256) void split_kernel(const float* __restrict__ X,
                                                    unsigned short* __restrict__ Ahi,
                                                    unsigned short* __restrict__ Alo,
                                                    unsigned short* __restrict__ BhiT,
                                                    unsigned short* __restrict__ BloT) {
    __shared__ unsigned short tileH[SPLIT_T * SPLIT_PAD];
    __shared__ unsigned short tileL[SPLIT_T * SPLIT_PAD];
    const int tid = threadIdx.x;
    const int k0  = blockIdx.x * SPLIT_T;
    const int i0  = blockIdx.y * SPLIT_T;

#pragma unroll
    for (int s = 0; s < 4; ++s) {
        const int e  = tid + s * 256;  // float4 index in 64x64 tile (16 per row)
        const int r  = e >> 4;
        const int c4 = e & 15;
        float4 x = *(const float4*)(X + (size_t)(i0 + r) * K_DIM + k0 + c4 * 4);
        unsigned short h0, h1, h2, h3, l0, l1, l2, l3;
        split_bf16(x.x, h0, l0); split_bf16(x.y, h1, l1);
        split_bf16(x.z, h2, l2); split_bf16(x.w, h3, l3);
        ushort4 hv = make_ushort4(h0, h1, h2, h3);
        ushort4 lv = make_ushort4(l0, l1, l2, l3);
        const size_t go = (size_t)(i0 + r) * K_DIM + k0 + c4 * 4;
        *(ushort4*)&Ahi[go] = hv;
        *(ushort4*)&Alo[go] = lv;
        *(ushort4*)&tileH[r * SPLIT_PAD + c4 * 4] = hv;
        *(ushort4*)&tileL[r * SPLIT_PAD + c4 * 4] = lv;
    }
    __syncthreads();
#pragma unroll
    for (int s = 0; s < 4; ++s) {
        const int e  = tid + s * 256;  // 64 k-rows x 16 i-chunks
        const int kk = e >> 4;
        const int i4 = e & 15;
        ushort4 hv = make_ushort4(tileH[(i4 * 4 + 0) * SPLIT_PAD + kk],
                                  tileH[(i4 * 4 + 1) * SPLIT_PAD + kk],
                                  tileH[(i4 * 4 + 2) * SPLIT_PAD + kk],
                                  tileH[(i4 * 4 + 3) * SPLIT_PAD + kk]);
        ushort4 lv = make_ushort4(tileL[(i4 * 4 + 0) * SPLIT_PAD + kk],
                                  tileL[(i4 * 4 + 1) * SPLIT_PAD + kk],
                                  tileL[(i4 * 4 + 2) * SPLIT_PAD + kk],
                                  tileL[(i4 * 4 + 3) * SPLIT_PAD + kk]);
        const size_t go = (size_t)(k0 + kk) * N_ROWS + i0 + i4 * 4;
        *(ushort4*)&BhiT[go] = hv;
        *(ushort4*)&BloT[go] = lv;
    }
}

// ---------------------------------------------------------------------------
// Kernel 1b: exact fp32 row squared-norms + init of per-row max/min bits
// ---------------------------------------------------------------------------
__global__ __launch_bounds__(256) void prep_kernel(const float* __restrict__ X,
                                                   float* __restrict__ sq,
                                                   int* __restrict__ rowmax,
                                                   int* __restrict__ rowmin) {
    const int row = blockIdx.x;
    const float* xr = X + (size_t)row * K_DIM;
    float s = 0.0f;
    for (int c = threadIdx.x; c < K_DIM; c += 256) {
        float v = xr[c];
        s = fmaf(v, v, s);
    }
#pragma unroll
    for (int m = 16; m >= 1; m >>= 1) s += __shfl_xor(s, m, 32);
    __shared__ float red[8];
    if ((threadIdx.x & 31) == 0) red[threadIdx.x >> 5] = s;
    __syncthreads();
    if (threadIdx.x == 0) {
        float t = 0.0f;
#pragma unroll
        for (int w = 0; w < 8; ++w) t += red[w];
        sq[row]     = t;
        rowmax[row] = 0;           // dist >= 0; diag always positive -> safe floor
        rowmin[row] = 0x7F7F7FFF;  // large finite f32 bit pattern (~3.396e38, just below FLT_MAX)
    }
}

// ---------------------------------------------------------------------------
// Kernel 2: bf16x3 WMMA Gram tiles + fused triplet-mining epilogue.
// Grid (32,32); 256 threads = 8 wave32 (wm 0..3 x wn 0..1), 2x4 16x16 accs/wave.
// Double-buffered LDS, one barrier per iteration. Slice k+1 is staged fully
// (batched distinct-register loads, then batched LDS stores) BEFORE the WMMAs
// of slice k so the copy registers die early (no spills); the staging stall is
// hidden by cross-wave overlap (74 KB LDS -> 4 workgroups per WGP).
// ---------------------------------------------------------------------------
__global__ __launch_bounds__(256) void gram_kernel(const unsigned short* __restrict__ gAhi,
                                                   const unsigned short* __restrict__ gAlo,
                                                   const unsigned short* __restrict__ gBhiT,
                                                   const unsigned short* __restrict__ gBloT,
                                                   const long long* __restrict__ tgt,
                                                   const float* __restrict__ sq,
                                                   int* __restrict__ rowmax,
                                                   int* __restrict__ rowmin) {
    __shared__ unsigned short AsHi[2][TM * AS_STRIDE];
    __shared__ unsigned short AsLo[2][TM * AS_STRIDE];
    __shared__ unsigned short BsHi[2][TK * BS_STRIDE];  // Bs[k][j]
    __shared__ unsigned short BsLo[2][TK * BS_STRIDE];

    const int tid  = threadIdx.x;
    const int lane = tid & 31;
    const int wave = tid >> 5;
    const int wm   = wave >> 1;  // 0..3
    const int wn   = wave & 1;   // 0..1
    const int i0   = blockIdx.y * TM;
    const int j0   = blockIdx.x * TN;

    // Copy-chunk geometry (k-invariant): 2 chunks per thread per array.
    const int e0 = tid, e1 = tid + 256;
    const int r0 = e0 >> 2,  q0 = e0 & 3;     // A: 128 rows x 4 chunks
    const int r1 = e1 >> 2,  q1 = e1 & 3;
    const int kk0 = e0 >> 4, qb0 = e0 & 15;   // B: 32 k-rows x 16 chunks
    const int kk1 = e1 >> 4, qb1 = e1 & 15;
    const size_t baseA0 = (size_t)(i0 + r0) * K_DIM + q0 * 8;
    const size_t baseA1 = (size_t)(i0 + r1) * K_DIM + q1 * 8;
    const size_t baseB0 = (size_t)kk0 * N_ROWS + j0 + qb0 * 8;
    const size_t baseB1 = (size_t)kk1 * N_ROWS + j0 + qb1 * 8;
    const int dA0 = r0 * AS_STRIDE + q0 * 8;
    const int dA1 = r1 * AS_STRIDE + q1 * 8;
    const int dB0 = kk0 * BS_STRIDE + qb0 * 8;
    const int dB1 = kk1 * BS_STRIDE + qb1 * 8;

    // Stage one 32-wide k-slice into LDS buffer `buf`: 8 loads into distinct
    // VGPRs (clause, staggered waits), then 8 stores. Short live range.
    auto stage_slice = [&](int kb, int buf) {
        const size_t ob = (size_t)kb * N_ROWS;
        uint4 ld0 = *(const uint4*)&gAhi[baseA0 + kb];
        uint4 ld1 = *(const uint4*)&gAlo[baseA0 + kb];
        uint4 ld2 = *(const uint4*)&gAhi[baseA1 + kb];
        uint4 ld3 = *(const uint4*)&gAlo[baseA1 + kb];
        uint4 ld4 = *(const uint4*)&gBhiT[baseB0 + ob];
        uint4 ld5 = *(const uint4*)&gBloT[baseB0 + ob];
        uint4 ld6 = *(const uint4*)&gBhiT[baseB1 + ob];
        uint4 ld7 = *(const uint4*)&gBloT[baseB1 + ob];
        *(uint4*)&AsHi[buf][dA0] = ld0;
        *(uint4*)&AsLo[buf][dA0] = ld1;
        *(uint4*)&AsHi[buf][dA1] = ld2;
        *(uint4*)&AsLo[buf][dA1] = ld3;
        *(uint4*)&BsHi[buf][dB0] = ld4;
        *(uint4*)&BsLo[buf][dB0] = ld5;
        *(uint4*)&BsHi[buf][dB1] = ld6;
        *(uint4*)&BsLo[buf][dB1] = ld7;
    };

    v8f acc[2][4];
#pragma unroll
    for (int mt = 0; mt < 2; ++mt)
#pragma unroll
        for (int nt = 0; nt < 4; ++nt)
#pragma unroll
            for (int e = 0; e < 8; ++e) acc[mt][nt][e] = 0.0f;

    // prologue: stage slice 0
    stage_slice(0, 0);
    __syncthreads();

    int cur = 0;
    for (int k0 = 0; k0 < K_DIM; k0 += TK) {
        const bool more = (k0 + TK < K_DIM);
        if (more) {
            stage_slice(k0 + TK, cur ^ 1);  // fill other buffer before computing
            if (k0 + 2 * TK < K_DIM) {      // prefetch slice k+2 -> global_prefetch_b8
                __builtin_prefetch(&gAhi[baseA0 + k0 + 2 * TK], 0, 1);
                __builtin_prefetch(&gAlo[baseA1 + k0 + 2 * TK], 0, 1);
                __builtin_prefetch(&gBhiT[baseB0 + (size_t)(k0 + 2 * TK) * N_ROWS], 0, 1);
                __builtin_prefetch(&gBloT[baseB1 + (size_t)(k0 + 2 * TK) * N_ROWS], 0, 1);
            }
        }

        // ---- A fragments (lane = M; K halves split at lane 16) ----
        BF16Frag ahi[2], alo[2];
        const int mrow = wm * 32 + (lane & 15);
        const int koff = (lane >> 4) * 8;
#pragma unroll
        for (int mt = 0; mt < 2; ++mt) {
            const unsigned short* ph = &AsHi[cur][(mrow + mt * 16) * AS_STRIDE];
            const unsigned short* pl = &AsLo[cur][(mrow + mt * 16) * AS_STRIDE];
            ahi[mt].u[0] = *(const uint4*)(ph + koff);
            ahi[mt].u[1] = *(const uint4*)(ph + 16 + koff);
            alo[mt].u[0] = *(const uint4*)(pl + koff);
            alo[mt].u[1] = *(const uint4*)(pl + 16 + koff);
        }
        // ---- B fragments (lane = N; K halves split at lane 16, same k order as A) ----
        BF16Frag bhi[4], blo[4];
#pragma unroll
        for (int nt = 0; nt < 4; ++nt) {
            const int cb = wn * 64 + nt * 16;
            const int kb = (lane >> 4) * 8;
            const unsigned short* ph = &BsHi[cur][cb + (lane & 15)];
            const unsigned short* pl = &BsLo[cur][cb + (lane & 15)];
            unsigned short* bh = (unsigned short*)&bhi[nt];
            unsigned short* bl = (unsigned short*)&blo[nt];
#pragma unroll
            for (int s = 0; s < 16; ++s) {
                const int k = kb + s + ((s >> 3) << 3);
                bh[s] = ph[k * BS_STRIDE];
                bl[s] = pl[k * BS_STRIDE];
            }
        }

        // ---- bf16x3: hi*hi + hi*lo + lo*hi ----
#pragma unroll
        for (int mt = 0; mt < 2; ++mt)
#pragma unroll
            for (int nt = 0; nt < 4; ++nt) {
                acc[mt][nt] = __builtin_amdgcn_wmma_f32_16x16x32_bf16(
                    false, ahi[mt].v, false, bhi[nt].v, (short)0, acc[mt][nt], false, false);
                acc[mt][nt] = __builtin_amdgcn_wmma_f32_16x16x32_bf16(
                    false, ahi[mt].v, false, blo[nt].v, (short)0, acc[mt][nt], false, false);
                acc[mt][nt] = __builtin_amdgcn_wmma_f32_16x16x32_bf16(
                    false, alo[mt].v, false, bhi[nt].v, (short)0, acc[mt][nt], false, false);
            }

        __syncthreads();
        cur ^= 1;
    }

    // ---- fused epilogue: dist + masked hardest-pos/neg mining ----
    float sqj[4]; int tj[4];
#pragma unroll
    for (int nt = 0; nt < 4; ++nt) {
        const int j = j0 + wn * 64 + nt * 16 + (lane & 15);
        sqj[nt] = sq[j];
        tj[nt]  = (int)tgt[j];
    }
    const int rbase = i0 + wm * 32 + ((lane >> 4) << 3);
#pragma unroll
    for (int mt = 0; mt < 2; ++mt) {
#pragma unroll
        for (int r = 0; r < 8; ++r) {
            const int i = rbase + mt * 16 + r;
            const float sqi = sq[i];
            const int   ti  = (int)tgt[i];
            float ap = 0.0f;
            float an = 3.402823466e+38f;
#pragma unroll
            for (int nt = 0; nt < 4; ++nt) {
                float g    = acc[mt][nt][r];
                float d2   = sqi + sqj[nt] - 2.0f * g;
                float dist = sqrtf(fmaxf(d2, 1e-12f));
                if (ti == tj[nt]) ap = fmaxf(ap, dist);
                else              an = fminf(an, dist);
            }
#pragma unroll
            for (int m = 8; m >= 1; m >>= 1) {  // xor<=8 stays inside 16-lane groups
                ap = fmaxf(ap, __shfl_xor(ap, m, 32));
                an = fminf(an, __shfl_xor(an, m, 32));
            }
            if ((lane & 15) == 0) {
                atomicMax(&rowmax[i], __float_as_int(ap));  // exact for dist >= 0
                atomicMin(&rowmin[i], __float_as_int(an));
            }
        }
    }
}

// ---------------------------------------------------------------------------
// Kernel 3: loss = mean(relu(margin + ap - an))
// ---------------------------------------------------------------------------
__global__ __launch_bounds__(256) void loss_kernel(const int* __restrict__ rowmax,
                                                   const int* __restrict__ rowmin,
                                                   float* __restrict__ out) {
    float s = 0.0f;
    for (int i = threadIdx.x; i < N_ROWS; i += 256) {
        float ap = __int_as_float(rowmax[i]);
        float an = __int_as_float(rowmin[i]);
        s += fmaxf(0.0f, 0.3f + ap - an);
    }
#pragma unroll
    for (int m = 16; m >= 1; m >>= 1) s += __shfl_xor(s, m, 32);
    __shared__ float red[8];
    if ((threadIdx.x & 31) == 0) red[threadIdx.x >> 5] = s;
    __syncthreads();
    if (threadIdx.x == 0) {
        float t = 0.0f;
#pragma unroll
        for (int w = 0; w < 8; ++w) t += red[w];
        out[0] = t * (1.0f / (float)N_ROWS);
    }
}

extern "C" void kernel_launch(void* const* d_in, const int* in_sizes, int n_in,
                              void* d_out, int out_size, void* d_ws, size_t ws_size,
                              hipStream_t stream) {
    (void)in_sizes; (void)n_in; (void)out_size; (void)ws_size;
    const float*     X   = (const float*)d_in[0];
    const long long* tgt = (const long long*)d_in[1];

    // Workspace layout (needs ~64.1 MB):
    //   [0, 16KB)    sq        4096 f32
    //   [16, 32KB)   rowmax    4096 i32 (f32 bits)
    //   [32, 48KB)   rowmin    4096 i32 (f32 bits)
    //   [64KB ...)   Ahi, Alo (NxK u16), BhiT, BloT (KxN u16), 16 MB each
    float* sq     = (float*)d_ws;
    int*   rowmax = (int*)d_ws + N_ROWS;
    int*   rowmin = (int*)d_ws + 2 * N_ROWS;
    unsigned short* Ahi  = (unsigned short*)((char*)d_ws + 65536);
    unsigned short* Alo  = Ahi  + (size_t)N_ROWS * K_DIM;
    unsigned short* BhiT = Alo  + (size_t)N_ROWS * K_DIM;
    unsigned short* BloT = BhiT + (size_t)N_ROWS * K_DIM;

    split_kernel<<<dim3(K_DIM / SPLIT_T, N_ROWS / SPLIT_T), 256, 0, stream>>>(X, Ahi, Alo, BhiT, BloT);
    prep_kernel<<<N_ROWS, 256, 0, stream>>>(X, sq, rowmax, rowmin);
    gram_kernel<<<dim3(N_ROWS / TN, N_ROWS / TM), 256, 0, stream>>>(Ahi, Alo, BhiT, BloT,
                                                                    tgt, sq, rowmax, rowmin);
    loss_kernel<<<1, 256, 0, stream>>>(rowmax, rowmin, (float*)d_out);
}
